// SparseAutoencoder_86955907875024
// MI455X (gfx1250) — matrix operand order repaired and checked by reading, not compiled, not run on hardware
//
#include <hip/hip_runtime.h>

#define D_MODEL  2048
#define D_HIDDEN 8192
#define NROWS    8192
#define TOPK     100

typedef __attribute__((ext_vector_type(16))) __bf16 v16bf;
typedef __attribute__((ext_vector_type(8)))  float  v8f;

// ---------------------------------------------------------------------------
// Kernel 1: pre = (x - b_pre) @ W_enc + b_enc   [8192 x 8192], bf16 WMMA
// Block = 256 threads (8 waves). Tile 128x128, BK=32.
// Wave (wr,wc) = (w>>2, w&3) computes a 64x32 region = 4x2 subtiles of 16x16.
// ---------------------------------------------------------------------------
__global__ __launch_bounds__(256) void encoder_gemm_kernel(
    const float* __restrict__ x, const float* __restrict__ Wenc,
    const float* __restrict__ benc, const float* __restrict__ bpre,
    float* __restrict__ pre)
{
    __shared__ __bf16 As[128][34];   // [m][k], padded
    __shared__ __bf16 Bs[128][34];   // [n][k], padded (B stored by column)

    const int tid  = threadIdx.x;
    const int bm   = blockIdx.y * 128;
    const int bn   = blockIdx.x * 128;
    const int wave = tid >> 5;
    const int lane = tid & 31;
    const int wr   = wave >> 2;      // 0..1
    const int wc   = wave & 3;       // 0..3
    const int lm   = lane & 15;
    const int half = lane >> 4;

    v8f acc[4][2];
#pragma unroll
    for (int mi = 0; mi < 4; ++mi)
#pragma unroll
        for (int ni = 0; ni < 2; ++ni)
            acc[mi][ni] = (v8f){0.f,0.f,0.f,0.f,0.f,0.f,0.f,0.f};

    const int kxA = tid & 31;        // A loader: k within tile
    const int m0A = tid >> 5;        // A loader: starting m (8 rows/pass)
    const int nxB = tid & 127;       // B loader: n within tile
    const int k0B = tid >> 7;        // B loader: starting k (2 ks/pass)

    for (int k0 = 0; k0 < D_MODEL; k0 += 32) {
        // --- prefetch next K tiles (L2-resident streams; covers latency) ---
        if (k0 + 32 < D_MODEL) {
            __builtin_prefetch(&x[(size_t)(bm + m0A) * D_MODEL + k0 + 32 + kxA], 0, 0);
            __builtin_prefetch(&Wenc[(size_t)(k0 + 32 + k0B) * D_HIDDEN + bn + nxB], 0, 0);
        }
        // --- stage A tile (128x32 fp32 -> bf16, subtract b_pre) ---
        {
            const float bp = bpre[k0 + kxA];
#pragma unroll
            for (int it = 0; it < 16; ++it) {
                const int m = m0A + it * 8;
                As[m][kxA] = (__bf16)(x[(size_t)(bm + m) * D_MODEL + k0 + kxA] - bp);
            }
        }
        // --- stage B tile (32x128 fp32 -> bf16, transposed to [n][k]) ---
        {
#pragma unroll
            for (int it = 0; it < 16; ++it) {
                const int k = k0B + it * 2;
                Bs[nxB][k] = (__bf16)Wenc[(size_t)(k0 + k) * D_HIDDEN + bn + nxB];
            }
        }
        __syncthreads();

        // --- build B fragments (layout: VGPR vr holds K=2vr,2vr+1 (+16 upper lanes)) ---
        v16bf bfrag[2];
#pragma unroll
        for (int ni = 0; ni < 2; ++ni) {
            const int n = wc * 32 + ni * 16 + lm;
#pragma unroll
            for (int vr = 0; vr < 8; ++vr) {
                const int kb = 2 * vr + 16 * half;
                bfrag[ni][2 * vr]     = Bs[n][kb];
                bfrag[ni][2 * vr + 1] = Bs[n][kb + 1];
            }
        }
        // --- A fragments + WMMA ---
#pragma unroll
        for (int mi = 0; mi < 4; ++mi) {
            v16bf afrag;
            const int m = wr * 64 + mi * 16 + lm;
#pragma unroll
            for (int vr = 0; vr < 8; ++vr) {
                // this code's contiguous k order (same as B): VGPR vr holds K=2vr,2vr+1 (+16 upper lanes)
                const int kb = 2 * vr + 16 * half;
                afrag[2 * vr]     = As[m][kb];
                afrag[2 * vr + 1] = As[m][kb + 1];
            }
#pragma unroll
            for (int ni = 0; ni < 2; ++ni) {
                acc[mi][ni] = __builtin_amdgcn_wmma_f32_16x16x32_bf16(
                    false, afrag, false, bfrag[ni], (short)0, acc[mi][ni], false, false);
            }
        }
        __syncthreads();
    }

    // --- store D + b_enc ; C/D layout: VGPR r -> (m = r + 8*half, n = lane&15) ---
#pragma unroll
    for (int mi = 0; mi < 4; ++mi)
#pragma unroll
        for (int ni = 0; ni < 2; ++ni) {
            const int n  = bn + wc * 32 + ni * 16 + lm;
            const float be = benc[n];
#pragma unroll
            for (int r = 0; r < 8; ++r) {
                const int m = bm + wr * 64 + mi * 16 + r + half * 8;
                pre[(size_t)m * D_HIDDEN + n] = acc[mi][ni][r] + be;
            }
        }
}

// ---------------------------------------------------------------------------
// Kernel 2: per-row top-200 (ranks 1-100 = topk, ranks 101-200 = aux topk),
// then overwrite the pre row in-place with the sparse ReLU'd hidden row.
// One block per row, 256 threads, 32 elements/thread in registers.
// Wave32 butterfly (shfl_xor) intra-wave; only 2 barriers per iteration.
// ---------------------------------------------------------------------------
__global__ __launch_bounds__(256) void topk_kernel(
    float* __restrict__ pre,            // in: pre, out: hidden (in-place)
    int* __restrict__ tk_idx, float* __restrict__ tk_val,
    int* __restrict__ ax_idx, float* __restrict__ ax_val,
    float* __restrict__ l0_row)
{
    const int row  = blockIdx.x;
    const int tid  = threadIdx.x;
    const int lane = tid & 31;
    const int wave = tid >> 5;
    float* prow = pre + (size_t)row * D_HIDDEN;

    float v[32];
#pragma unroll
    for (int j = 0; j < 32; ++j) v[j] = prow[j * 256 + tid];

    float lmax = v[0]; int lj = 0;
#pragma unroll
    for (int j = 1; j < 32; ++j) if (v[j] > lmax) { lmax = v[j]; lj = j; }
    int lidx = lj * 256 + tid;

    __shared__ float sv[8];
    __shared__ int   si[8];

    for (int it = 0; it < 2 * TOPK; ++it) {
        // intra-wave butterfly argmax (lower index wins ties) -> all lanes agree
        float wvv = lmax; int wii = lidx;
#pragma unroll
        for (int off = 16; off > 0; off >>= 1) {
            const float ov = __shfl_xor(wvv, off, 32);
            const int   oi = __shfl_xor(wii, off, 32);
            if (ov > wvv || (ov == wvv && oi < wii)) { wvv = ov; wii = oi; }
        }
        if (lane == 0) { sv[wave] = wvv; si[wave] = wii; }
        __syncthreads();
        // every thread serially reduces 8 wave maxima (broadcast LDS reads);
        // commutative max-with-min-index => all threads agree on the winner
        float bw = sv[0]; int bi = si[0];
#pragma unroll
        for (int w = 1; w < 8; ++w) {
            const float ov = sv[w]; const int oi = si[w];
            if (ov > bw || (ov == bw && oi < bi)) { bw = ov; bi = oi; }
        }
        __syncthreads();   // protect sv/si before next iteration's write

        if (tid == 0) {
            if (it < TOPK) { tk_idx[row * TOPK + it] = bi; tk_val[row * TOPK + it] = bw; }
            else           { ax_idx[row * TOPK + it - TOPK] = bi; ax_val[row * TOPK + it - TOPK] = bw; }
        }
        if ((bi & 255) == tid) {
            v[bi >> 8] = -__builtin_inff();
            lmax = v[0]; lj = 0;
#pragma unroll
            for (int j = 1; j < 32; ++j) if (v[j] > lmax) { lmax = v[j]; lj = j; }
            lidx = lj * 256 + tid;
        }
    }

    // overwrite row with sparse hidden = scatter(relu(topk_vals))
#pragma unroll
    for (int j = 0; j < 32; ++j) prow[j * 256 + tid] = 0.0f;
    __syncthreads();
    __shared__ int cnt;
    if (tid == 0) cnt = 0;
    __syncthreads();
    if (tid < TOPK) {
        const float val = tk_val[row * TOPK + tid];
        const int   idx = tk_idx[row * TOPK + tid];
        const float rv = val > 0.f ? val : 0.f;
        prow[idx] = rv;
        if (rv > 0.f) atomicAdd(&cnt, 1);
    }
    __syncthreads();
    if (tid == 0) l0_row[row] = (float)cnt;
}

// ---------------------------------------------------------------------------
// Kernel 3: sparse decode for both main and aux reconstructions + per-row SSEs.
// One block per row; thread t owns cols t, t+256, ..., t+1792.
// ---------------------------------------------------------------------------
__global__ __launch_bounds__(256) void decode_kernel(
    const float* __restrict__ x, const float* __restrict__ Wdec,
    const float* __restrict__ bdec, const float* __restrict__ bpre,
    const int* __restrict__ tk_idx, const float* __restrict__ tk_val,
    const int* __restrict__ ax_idx, const float* __restrict__ ax_val,
    float* __restrict__ recon, float* __restrict__ sse_r_row, float* __restrict__ sse_a_row)
{
    const int row = blockIdx.x;
    const int tid = threadIdx.x;

    __shared__ int   fi[TOPK]; __shared__ float fv[TOPK];
    __shared__ int   ai[TOPK]; __shared__ float av[TOPK];
    if (tid < TOPK) {
        fi[tid] = tk_idx[row * TOPK + tid];
        const float v = tk_val[row * TOPK + tid]; fv[tid] = v > 0.f ? v : 0.f;
        ai[tid] = ax_idx[row * TOPK + tid];
        const float a = ax_val[row * TOPK + tid]; av[tid] = a > 0.f ? a : 0.f;
    }
    __syncthreads();

    float accr[8] = {0}, acca[8] = {0};
    for (int f = 0; f < TOPK; ++f) {
        const float* wrp = Wdec + (size_t)fi[f] * D_MODEL;
        const float* wap = Wdec + (size_t)ai[f] * D_MODEL;
        const float vf = fv[f], va = av[f];
#pragma unroll
        for (int c = 0; c < 8; ++c) {
            const int col = tid + c * 256;
            accr[c] = fmaf(vf, wrp[col], accr[c]);
            acca[c] = fmaf(va, wap[col], acca[c]);
        }
    }

    float sr = 0.f, sa = 0.f;
#pragma unroll
    for (int c = 0; c < 8; ++c) {
        const int col = tid + c * 256;
        const float base = bdec[col] + bpre[col];
        const float rec  = accr[c] + base;
        recon[(size_t)row * D_MODEL + col] = rec;
        const float xv = x[(size_t)row * D_MODEL + col];
        const float dr = rec - xv;        sr = fmaf(dr, dr, sr);
        const float resid = xv - rec;
        const float auxr  = acca[c] + base;
        const float da = auxr - resid;    sa = fmaf(da, da, sa);
    }

    // deterministic block reduction (wave butterfly + small LDS combine)
#pragma unroll
    for (int off = 16; off > 0; off >>= 1) {
        sr += __shfl_xor(sr, off, 32);
        sa += __shfl_xor(sa, off, 32);
    }
    __shared__ float rr[8], ra[8];
    if ((tid & 31) == 0) { rr[tid >> 5] = sr; ra[tid >> 5] = sa; }
    __syncthreads();
    if (tid == 0) {
        float tr = 0.f, ta = 0.f;
#pragma unroll
        for (int w = 0; w < 8; ++w) { tr += rr[w]; ta += ra[w]; }
        sse_r_row[row] = tr;
        sse_a_row[row] = ta;
    }
}

// ---------------------------------------------------------------------------
// Kernel 4: deterministic final reduction of per-row partials -> 4 scalars.
// ---------------------------------------------------------------------------
__global__ __launch_bounds__(256) void finalize_kernel(
    const float* __restrict__ sse_r_row, const float* __restrict__ sse_a_row,
    const float* __restrict__ l0_row, float* __restrict__ outs)
{
    const int tid = threadIdx.x;
    float sr = 0.f, sa = 0.f, sl = 0.f;
    for (int i = tid; i < NROWS; i += 256) { sr += sse_r_row[i]; sa += sse_a_row[i]; sl += l0_row[i]; }
    __shared__ float r0[256], r1[256], r2[256];
    r0[tid] = sr; r1[tid] = sa; r2[tid] = sl; __syncthreads();
    for (int s = 128; s > 0; s >>= 1) {
        if (tid < s) { r0[tid] += r0[tid + s]; r1[tid] += r1[tid + s]; r2[tid] += r2[tid + s]; }
        __syncthreads();
    }
    if (tid == 0) {
        const float denom = (float)NROWS * (float)D_MODEL;
        const float rl = r0[0] / denom;
        const float al = (r1[0] / denom) * (1.0f / 32.0f);
        outs[0] = rl + al;              // loss
        outs[1] = rl;                   // reconstruction_loss
        outs[2] = al;                   // aux_loss
        outs[3] = r2[0] / (float)NROWS; // l0_sparsity
    }
}

// ---------------------------------------------------------------------------
extern "C" void kernel_launch(void* const* d_in, const int* in_sizes, int n_in,
                              void* d_out, int out_size, void* d_ws, size_t ws_size,
                              hipStream_t stream)
{
    const float* x    = (const float*)d_in[0];
    const float* Wenc = (const float*)d_in[1];
    const float* benc = (const float*)d_in[2];
    const float* Wdec = (const float*)d_in[3];
    const float* bdec = (const float*)d_in[4];
    const float* bpre = (const float*)d_in[5];

    float* out     = (float*)d_out;
    float* recon   = out;                                   // [N, D]
    float* hidden  = out + (size_t)NROWS * D_MODEL;         // [N, H] (pre, then sparse)
    float* scalars = hidden + (size_t)NROWS * D_HIDDEN;     // 4 scalars

    char* ws = (char*)d_ws;
    int*   tk_idx = (int*)ws;            ws += sizeof(int)   * (size_t)NROWS * TOPK;
    float* tk_val = (float*)ws;          ws += sizeof(float) * (size_t)NROWS * TOPK;
    int*   ax_idx = (int*)ws;            ws += sizeof(int)   * (size_t)NROWS * TOPK;
    float* ax_val = (float*)ws;          ws += sizeof(float) * (size_t)NROWS * TOPK;
    float* sse_r_row = (float*)ws;       ws += sizeof(float) * NROWS;
    float* sse_a_row = (float*)ws;       ws += sizeof(float) * NROWS;
    float* l0_row    = (float*)ws;       ws += sizeof(float) * NROWS;

    dim3 gemm_grid(D_HIDDEN / 128, NROWS / 128);
    encoder_gemm_kernel<<<gemm_grid, dim3(256), 0, stream>>>(x, Wenc, benc, bpre, hidden);
    topk_kernel<<<dim3(NROWS), dim3(256), 0, stream>>>(hidden, tk_idx, tk_val, ax_idx, ax_val, l0_row);
    decode_kernel<<<dim3(NROWS), dim3(256), 0, stream>>>(x, Wdec, bdec, bpre,
                                                         tk_idx, tk_val, ax_idx, ax_val,
                                                         recon, sse_r_row, sse_a_row);
    finalize_kernel<<<dim3(1), dim3(256), 0, stream>>>(sse_r_row, sse_a_row, l0_row, scalars);
}
